// Cross_SA_Layer_Linear_80882824119001
// MI455X (gfx1250) — matrix operand order repaired and checked by reading, not compiled, not run on hardware
//
#include <hip/hip_runtime.h>

// ---------------------------------------------------------------------------
// MI455X (gfx1250, wave32) implementation.
// All GEMMs use V_WMMA_F32_16X16X32_BF16. Each wave owns a 32x32 output
// macro-tile: 2 A-tiles x 2 B-tiles -> 4 f32 accumulators per K-step, i.e.
// 4 WMMAs per 8 global_load_b128 (2x the register-level operand reuse of a
// single 16x16 tile). Problem is HBM-bandwidth bound (~0.5 GB traffic @
// 23.3 TB/s), so activations/weights are converted to bf16 once and tensor
// layouts are chosen so every WMMA operand load is contiguous per lane.
// ---------------------------------------------------------------------------

typedef __attribute__((ext_vector_type(16))) __bf16 v16bf;
typedef __attribute__((ext_vector_type(8)))  __bf16 v8bf;
typedef __attribute__((ext_vector_type(8)))  float  v8f;
typedef __bf16 bf16_t;

#define C_DIM  512
#define N_DIM  4096
#define BB     4        // batches in "first"
#define M_REST 12       // batches in "rest"
#define CG     128      // channels per shared group
#define H_DIM  2048     // 4*C
#define CT32   (C_DIM/32)   // 16
#define NT32   (N_DIM/32)   // 128
#define HT32   (H_DIM/32)   // 64

// ---------------- WMMA tile helpers (ISA 7.12.2 layouts) -------------------

// A (16x32 bf16, MxK): lanes 0-15 -> rows M=0..15, K in {h*8..h*8+7, 16+h*8..}
static __device__ __forceinline__ v16bf load_a16(const bf16_t* A, int lda,
                                                 int k0, int lane) {
  const int m = lane & 15, h = lane >> 4;
  const bf16_t* p = A + (size_t)m * lda + k0 + h * 8;
  v8bf a0 = *(const v8bf*)(p);        // K = k0+h*8 .. +7
  v8bf a1 = *(const v8bf*)(p + 16);   // K = k0+16+h*8 .. +7
  return __builtin_shufflevector(a0, a1, 0,1,2,3,4,5,6,7,8,9,10,11,12,13,14,15);
}

// B (32x16 bf16, KxN): lane = column N (lane&15), K in {h*8..h*8+7, 16+h*8..}
// (same order as A).  Memory layout: B[col*ldb + k], K contiguous.
static __device__ __forceinline__ v16bf load_b16(const bf16_t* B, int ldb,
                                                 int k0, int lane) {
  const int c = lane & 15, h = lane >> 4;
  const bf16_t* p = B + (size_t)c * ldb + k0 + h * 8;
  v8bf b0 = *(const v8bf*)(p);
  v8bf b1 = *(const v8bf*)(p + 16);
  return __builtin_shufflevector(b0, b1, 0,1,2,3,4,5,6,7,8,9,10,11,12,13,14,15);
}

static __device__ __forceinline__ v8f wmma_bf16(v16bf a, v16bf b, v8f c) {
  return __builtin_amdgcn_wmma_f32_16x16x32_bf16(false, a, false, b,
                                                 (short)0, c, false, false);
}

// 32x32 macro-tile: rows {0,16}+A, cols {0,16}+B. acc[ti*2+tj].
static __device__ __forceinline__ void gemm_tile_2x2(const bf16_t* __restrict__ A, int lda,
                                                     const bf16_t* __restrict__ B, int ldb,
                                                     int K, int lane, v8f acc[4]) {
  for (int k0 = 0; k0 < K; k0 += 32) {
    if (k0 + 32 < K) {
      __builtin_prefetch((const void*)(A + (size_t)(lane & 15) * lda + k0 + 32), 0, 1);
      __builtin_prefetch((const void*)(A + (size_t)(16 + (lane & 15)) * lda + k0 + 32), 0, 1);
      __builtin_prefetch((const void*)(B + (size_t)(lane & 15) * ldb + k0 + 32), 0, 1);
      __builtin_prefetch((const void*)(B + (size_t)(16 + (lane & 15)) * ldb + k0 + 32), 0, 1);
    }
    v16bf a0 = load_a16(A, lda, k0, lane);
    v16bf a1 = load_a16(A + (size_t)16 * lda, lda, k0, lane);
    v16bf b0 = load_b16(B, ldb, k0, lane);
    v16bf b1 = load_b16(B + (size_t)16 * ldb, ldb, k0, lane);
    acc[0] = wmma_bf16(a0, b0, acc[0]);
    acc[1] = wmma_bf16(a0, b1, acc[1]);
    acc[2] = wmma_bf16(a1, b0, acc[2]);
    acc[3] = wmma_bf16(a1, b1, acc[3]);
  }
}

static __device__ __forceinline__ float block_reduce_sum(float v, float* sd) {
  const int tid = threadIdx.x;
  sd[tid] = v; __syncthreads();
  for (int s = blockDim.x >> 1; s > 0; s >>= 1) {
    if (tid < s) sd[tid] += sd[tid + s];
    __syncthreads();
  }
  float r = sd[0]; __syncthreads();
  return r;
}

// ---------------- elementwise / transpose kernels --------------------------

__global__ void k_cvt_bf16(const float* __restrict__ s, bf16_t* __restrict__ d, int n) {
  for (int i = blockIdx.x * blockDim.x + threadIdx.x; i < n; i += gridDim.x * blockDim.x)
    d[i] = (bf16_t)s[i];
}

// x (16,C,N) f32  ->  xbT (16,N,C) bf16   (LDS-tiled transpose + convert)
__global__ __launch_bounds__(256) void k_transpose_cvt(const float* __restrict__ x,
                                                       bf16_t* __restrict__ xbT) {
  __shared__ float tile[32][33];
  const int b  = blockIdx.z;
  const int c0 = blockIdx.x * 32;
  const int n0 = blockIdx.y * 32;
  const int tx = threadIdx.x;   // 0..31
  const int ty = threadIdx.y;   // 0..7
  #pragma unroll
  for (int i = 0; i < 4; ++i) {
    int c = c0 + ty + i * 8;
    tile[ty + i * 8][tx] = x[((size_t)b * C_DIM + c) * N_DIM + n0 + tx];
  }
  __syncthreads();
  #pragma unroll
  for (int i = 0; i < 4; ++i) {
    int n = n0 + ty + i * 8;
    xbT[((size_t)b * N_DIM + n) * C_DIM + c0 + tx] = (bf16_t)tile[tx][ty + i * 8];
  }
}

// ---------------- GEMM kernels (32x32 macro-tile per wave) -----------------

// q = phi(grouped(Wq, rest)) -> qt (12,N,C) bf16 ; k0 = phi(grouped(Wk, first)) -> kb (4,C,N)
__global__ __launch_bounds__(256) void k_qk(const bf16_t* __restrict__ xbT,
                                            const bf16_t* __restrict__ Wq,
                                            const bf16_t* __restrict__ Wk,
                                            bf16_t* __restrict__ qt,
                                            bf16_t* __restrict__ kb) {
  const int lane = threadIdx.x & 31;
  const int tile = blockIdx.x * 8 + (threadIdx.x >> 5);
  const int tpb = CT32 * NT32;
  const int j  = tile / tpb;                // batch 0..15
  const int t  = tile % tpb;
  const int c0 = (t / NT32) * 32;           // output channel macro-tile
  const int n0 = (t % NT32) * 32;
  const int gi = c0 >> 7;                   // shared group (32 | 128)
  const bf16_t* A  = (j < BB ? Wk : Wq) + (size_t)gi * CG * CG
                   + (size_t)(c0 & (CG - 1)) * CG;
  const bf16_t* Bm = xbT + ((size_t)j * N_DIM + n0) * C_DIM + gi * CG;
  v8f acc[4] = {};
  gemm_tile_2x2(A, CG, Bm, C_DIM, CG, lane, acc);
  const int col = lane & 15, mb = (lane >> 4) * 8;
  if (j < BB) {            // k0 -> (b,C,N), phi = relu+1
    #pragma unroll
    for (int ti = 0; ti < 2; ++ti)
      #pragma unroll
      for (int tj = 0; tj < 2; ++tj) {
        bf16_t* dst = kb + ((size_t)j * C_DIM + c0 + ti * 16 + mb) * N_DIM
                         + n0 + tj * 16 + col;
        v8f a = acc[ti * 2 + tj];
        #pragma unroll
        for (int r = 0; r < 8; ++r) {
          float v = a[r]; v = (v > 0.f ? v : 0.f) + 1.f;
          dst[(size_t)r * N_DIM] = (bf16_t)v;
        }
      }
  } else {                 // q -> (m,N,C), contiguous 16B store per lane
    #pragma unroll
    for (int ti = 0; ti < 2; ++ti)
      #pragma unroll
      for (int tj = 0; tj < 2; ++tj) {
        v8f a = acc[ti * 2 + tj];
        v8bf pk;
        #pragma unroll
        for (int r = 0; r < 8; ++r) {
          float v = a[r]; v = (v > 0.f ? v : 0.f) + 1.f;
          pk[r] = (bf16_t)v;
        }
        *(v8bf*)(qt + ((size_t)(j - BB) * N_DIM + n0 + tj * 16 + col) * C_DIM
                    + c0 + ti * 16 + mb) = pk;
      }
  }
}

// v0 = Wv @ first -> vb (4,C,N) ; phi_all = Wphi @ x -> phib (16,N,C)
__global__ __launch_bounds__(256) void k_vphi(const bf16_t* __restrict__ xbT,
                                              const bf16_t* __restrict__ Wv,
                                              const bf16_t* __restrict__ Wp,
                                              bf16_t* __restrict__ vb,
                                              bf16_t* __restrict__ phib) {
  const int lane = threadIdx.x & 31;
  const int tile = blockIdx.x * 8 + (threadIdx.x >> 5);
  const int tpb = CT32 * NT32;
  const int j  = tile / tpb;                // 0..3 Wv(first), 4..19 Wphi(all)
  const int t  = tile % tpb;
  const int c0 = (t / NT32) * 32;
  const int n0 = (t % NT32) * 32;
  const int batch = (j < BB) ? j : (j - BB);
  const bf16_t* A  = (j < BB ? Wv : Wp) + (size_t)c0 * C_DIM;
  const bf16_t* Bm = xbT + ((size_t)batch * N_DIM + n0) * C_DIM;
  v8f acc[4] = {};
  gemm_tile_2x2(A, C_DIM, Bm, C_DIM, C_DIM, lane, acc);
  const int col = lane & 15, mb = (lane >> 4) * 8;
  if (j < BB) {
    #pragma unroll
    for (int ti = 0; ti < 2; ++ti)
      #pragma unroll
      for (int tj = 0; tj < 2; ++tj) {
        bf16_t* dst = vb + ((size_t)j * C_DIM + c0 + ti * 16 + mb) * N_DIM
                         + n0 + tj * 16 + col;
        v8f a = acc[ti * 2 + tj];
        #pragma unroll
        for (int r = 0; r < 8; ++r) dst[(size_t)r * N_DIM] = (bf16_t)a[r];
      }
  } else {
    #pragma unroll
    for (int ti = 0; ti < 2; ++ti)
      #pragma unroll
      for (int tj = 0; tj < 2; ++tj) {
        v8f a = acc[ti * 2 + tj];
        v8bf pk;
        #pragma unroll
        for (int r = 0; r < 8; ++r) pk[r] = (bf16_t)a[r];
        *(v8bf*)(phib + ((size_t)batch * N_DIM + n0 + tj * 16 + col) * C_DIM
                      + c0 + ti * 16 + mb) = pk;
      }
  }
}

// kv^T: kvt[b][d][c] = sum_n khat[b][c][n]*vb[b][d][n] / sqrt(C)   (bf16)
__global__ __launch_bounds__(256) void k_kv(const bf16_t* __restrict__ khat,
                                            const bf16_t* __restrict__ vb,
                                            bf16_t* __restrict__ kvt) {
  const int lane = threadIdx.x & 31;
  const int tile = blockIdx.x * 8 + (threadIdx.x >> 5);
  const int tpb = CT32 * CT32;
  const int b  = tile / tpb;
  const int t  = tile % tpb;
  const int c0 = (t / CT32) * 32;   // rows = c
  const int d0 = (t % CT32) * 32;   // cols = d
  const bf16_t* A  = khat + ((size_t)b * C_DIM + c0) * N_DIM;   // K = n
  const bf16_t* Bm = vb   + ((size_t)b * C_DIM + d0) * N_DIM;
  v8f acc[4] = {};
  gemm_tile_2x2(A, N_DIM, Bm, N_DIM, N_DIM, lane, acc);
  const int col = lane & 15, mb = (lane >> 4) * 8;
  const float sc = 0.044194173824159216f;   // 1/sqrt(512)
  #pragma unroll
  for (int ti = 0; ti < 2; ++ti)
    #pragma unroll
    for (int tj = 0; tj < 2; ++tj) {
      v8f a = acc[ti * 2 + tj];
      v8bf pk;
      #pragma unroll
      for (int r = 0; r < 8; ++r) pk[r] = (bf16_t)(a[r] * sc);
      *(v8bf*)(kvt + ((size_t)b * C_DIM + d0 + tj * 16 + col) * C_DIM
                   + c0 + ti * 16 + mb) = pk;
    }
}

// ymean[n][d] = (1/12) sum_m (sum_c kvt[b][d][c]*qt[m][n][c]) * z[m][n] * phib[4+m][n][d]
__global__ __launch_bounds__(256) void k_y(const bf16_t* __restrict__ kvt,
                                           const bf16_t* __restrict__ qt,
                                           const float* __restrict__ z,
                                           const bf16_t* __restrict__ phib,
                                           float* __restrict__ ymean) {
  const int lane = threadIdx.x & 31;
  const int tile = blockIdx.x * 8 + (threadIdx.x >> 5);
  const int d0 = (tile / NT32) * 32;
  const int n0 = (tile % NT32) * 32;
  const int col = lane & 15, mb = (lane >> 4) * 8;
  v8f ys[4] = {};
  for (int m = 0; m < M_REST; ++m) {
    const int b = m & 3;
    const bf16_t* A  = kvt + ((size_t)b * C_DIM + d0) * C_DIM;
    const bf16_t* Bm = qt  + ((size_t)m * N_DIM + n0) * C_DIM;
    v8f acc[4] = {};
    gemm_tile_2x2(A, C_DIM, Bm, C_DIM, C_DIM, lane, acc);
    #pragma unroll
    for (int tj = 0; tj < 2; ++tj) {
      float zv = z[(size_t)m * N_DIM + n0 + tj * 16 + col];
      #pragma unroll
      for (int ti = 0; ti < 2; ++ti) {
        v8bf ph = *(const v8bf*)(phib + ((size_t)(BB + m) * N_DIM + n0 + tj * 16 + col) * C_DIM
                                      + d0 + ti * 16 + mb);
        v8f a = acc[ti * 2 + tj];
        #pragma unroll
        for (int r = 0; r < 8; ++r) ys[ti * 2 + tj][r] += a[r] * zv * (float)ph[r];
      }
    }
  }
  #pragma unroll
  for (int ti = 0; ti < 2; ++ti)
    #pragma unroll
    for (int tj = 0; tj < 2; ++tj) {
      v8f o;
      #pragma unroll
      for (int r = 0; r < 8; ++r) o[r] = ys[ti * 2 + tj][r] * (1.0f / 12.0f);
      *(v8f*)(ymean + (size_t)(n0 + tj * 16 + col) * C_DIM + d0 + ti * 16 + mb) = o;
    }
}

// h = relu(out1 @ fc1^T + b1) -> (4,N,H) bf16
__global__ __launch_bounds__(256) void k_fc1(const bf16_t* __restrict__ fc1wb,
                                             const bf16_t* __restrict__ out1b,
                                             const float* __restrict__ bias,
                                             bf16_t* __restrict__ hbuf) {
  const int lane = threadIdx.x & 31;
  const int tile = blockIdx.x * 8 + (threadIdx.x >> 5);
  const int tpb = HT32 * NT32;
  const int j  = tile / tpb;
  const int t  = tile % tpb;
  const int o0 = (t / NT32) * 32;
  const int n0 = (t % NT32) * 32;
  const bf16_t* A  = fc1wb + (size_t)o0 * C_DIM;
  const bf16_t* Bm = out1b + ((size_t)j * N_DIM + n0) * C_DIM;
  v8f acc[4] = {};
  gemm_tile_2x2(A, C_DIM, Bm, C_DIM, C_DIM, lane, acc);
  const int col = lane & 15, mb = (lane >> 4) * 8;
  #pragma unroll
  for (int ti = 0; ti < 2; ++ti) {
    v8f bi = *(const v8f*)(bias + o0 + ti * 16 + mb);
    #pragma unroll
    for (int tj = 0; tj < 2; ++tj) {
      v8f a = acc[ti * 2 + tj];
      v8bf pk;
      #pragma unroll
      for (int r = 0; r < 8; ++r) {
        float v = a[r] + bi[r]; v = v > 0.f ? v : 0.f;
        pk[r] = (bf16_t)v;
      }
      *(v8bf*)(hbuf + ((size_t)j * N_DIM + n0 + tj * 16 + col) * H_DIM
                    + o0 + ti * 16 + mb) = pk;
    }
  }
}

// o2 = out1 + h @ fc2^T + b2 -> (4,N,C) f32
__global__ __launch_bounds__(256) void k_fc2(const bf16_t* __restrict__ fc2wb,
                                             const bf16_t* __restrict__ hbuf,
                                             const float* __restrict__ bias,
                                             const float* __restrict__ out1f,
                                             float* __restrict__ o2) {
  const int lane = threadIdx.x & 31;
  const int tile = blockIdx.x * 8 + (threadIdx.x >> 5);
  const int tpb = CT32 * NT32;
  const int j  = tile / tpb;
  const int t  = tile % tpb;
  const int c0 = (t / NT32) * 32;
  const int n0 = (t % NT32) * 32;
  const bf16_t* A  = fc2wb + (size_t)c0 * H_DIM;
  const bf16_t* Bm = hbuf  + ((size_t)j * N_DIM + n0) * H_DIM;
  v8f acc[4] = {};
  gemm_tile_2x2(A, H_DIM, Bm, H_DIM, H_DIM, lane, acc);
  const int col = lane & 15, mb = (lane >> 4) * 8;
  #pragma unroll
  for (int ti = 0; ti < 2; ++ti) {
    v8f bi = *(const v8f*)(bias + c0 + ti * 16 + mb);
    #pragma unroll
    for (int tj = 0; tj < 2; ++tj) {
      const size_t ridx = ((size_t)j * N_DIM + n0 + tj * 16 + col) * C_DIM
                        + c0 + ti * 16 + mb;
      v8f res = *(const v8f*)(out1f + ridx);
      v8f a = acc[ti * 2 + tj];
      v8f o;
      #pragma unroll
      for (int r = 0; r < 8; ++r) o[r] = res[r] + a[r] + bi[r];
      *(v8f*)(o2 + ridx) = o;
    }
  }
}

// ---------------- reductions / normalizations ------------------------------

// Qg[b][c] = mean_n kb[b][c][n]
__global__ __launch_bounds__(256) void k_qg(const bf16_t* __restrict__ kb,
                                            float* __restrict__ Qg) {
  __shared__ float sd[256];
  const int row = blockIdx.x;               // b*C + c
  const bf16_t* pr = kb + (size_t)row * N_DIM;
  float s = 0.f;
  for (int n = threadIdx.x; n < N_DIM; n += 256) s += (float)pr[n];
  float tot = block_reduce_sum(s, sd);
  if (threadIdx.x == 0) Qg[row] = tot * (1.0f / N_DIM);
}

// alpha[b][n] = softmax_n(sum_c Qg[b][c]*kb[b][c][n]) * N
__global__ __launch_bounds__(512) void k_alpha(const bf16_t* __restrict__ kb,
                                               const float* __restrict__ Qg,
                                               float* __restrict__ alpha) {
  __shared__ float sv[N_DIM];   // 16 KB
  __shared__ float qg[C_DIM];
  __shared__ float red[512];
  const int b = blockIdx.x, tid = threadIdx.x;
  for (int c = tid; c < C_DIM; c += 512) qg[c] = Qg[b * C_DIM + c];
  __syncthreads();
  for (int n = tid; n < N_DIM; n += 512) {
    float s = 0.f;
    const bf16_t* colp = kb + (size_t)b * C_DIM * N_DIM + n;
    for (int c = 0; c < C_DIM; ++c) s += qg[c] * (float)colp[(size_t)c * N_DIM];
    sv[n] = s;
  }
  __syncthreads();
  float mx = -3.4e38f;
  for (int n = tid; n < N_DIM; n += 512) mx = fmaxf(mx, sv[n]);
  red[tid] = mx; __syncthreads();
  for (int st = 256; st > 0; st >>= 1) {
    if (tid < st) red[tid] = fmaxf(red[tid], red[tid + st]);
    __syncthreads();
  }
  mx = red[0]; __syncthreads();
  float sum = 0.f;
  for (int n = tid; n < N_DIM; n += 512) {
    float e = __expf(sv[n] - mx); sv[n] = e; sum += e;
  }
  red[tid] = sum; __syncthreads();
  for (int st = 256; st > 0; st >>= 1) {
    if (tid < st) red[tid] += red[tid + st];
    __syncthreads();
  }
  sum = red[0]; __syncthreads();
  const float inv = (float)N_DIM / sum;
  for (int n = tid; n < N_DIM; n += 512)
    alpha[(size_t)b * N_DIM + n] = sv[n] * inv;
}

// khat = kb*alpha (bf16), ksum[b][c] = sum_n khat
__global__ __launch_bounds__(256) void k_khat(const bf16_t* __restrict__ kb,
                                              const float* __restrict__ alpha,
                                              bf16_t* __restrict__ khat,
                                              float* __restrict__ ksum) {
  __shared__ float sd[256];
  const int row = blockIdx.x;               // b*C + c
  const int b = row >> 9;
  const bf16_t* pr = kb + (size_t)row * N_DIM;
  bf16_t* po = khat + (size_t)row * N_DIM;
  const float* al = alpha + (size_t)b * N_DIM;
  float s = 0.f;
  for (int n = threadIdx.x; n < N_DIM; n += 256) {
    float v = (float)pr[n] * al[n];
    po[n] = (bf16_t)v;
    s += v;
  }
  float tot = block_reduce_sum(s, sd);
  if (threadIdx.x == 0) ksum[row] = tot;
}

// z[m][n] = 1/(sum_c qt[m][n][c]*ksum[b][c] + eps)  (one wave per (m,n))
__global__ __launch_bounds__(256) void k_z(const bf16_t* __restrict__ qt,
                                           const float* __restrict__ ksum,
                                           float* __restrict__ z) {
  const int wid  = blockIdx.x * 8 + (threadIdx.x >> 5);
  const int lane = threadIdx.x & 31;
  const int m = wid / N_DIM, n = wid % N_DIM, b = m & 3;
  const bf16_t* p = qt + ((size_t)m * N_DIM + n) * C_DIM;
  const float* ks = ksum + (size_t)b * C_DIM;
  float s = 0.f;
  for (int c = lane; c < C_DIM; c += 32) s += (float)p[c] * ks[c];
  for (int off = 16; off > 0; off >>= 1) s += __shfl_down(s, off, 32);
  if (lane == 0) z[(size_t)m * N_DIM + n] = 1.0f / (s + 1e-6f);
}

// LN1: u = x + ymean*phi_first ; out1 = ln(u)  -> f32 + bf16, layout (b,n,c)
__global__ __launch_bounds__(256) void k_ln1(const float* __restrict__ x,
                                             const float* __restrict__ ymean,
                                             const bf16_t* __restrict__ phib,
                                             const float* __restrict__ w,
                                             const float* __restrict__ bia,
                                             float* __restrict__ out1f,
                                             bf16_t* __restrict__ out1b) {
  __shared__ float sd[256];
  const int bn = blockIdx.x;
  const int b = bn >> 12, n = bn & (N_DIM - 1);
  const int tid = threadIdx.x;
  const int c0 = tid, c1 = tid + 256;
  float u0 = x[((size_t)b * C_DIM + c0) * N_DIM + n]
           + ymean[(size_t)n * C_DIM + c0]
           * (float)phib[((size_t)b * N_DIM + n) * C_DIM + c0];
  float u1 = x[((size_t)b * C_DIM + c1) * N_DIM + n]
           + ymean[(size_t)n * C_DIM + c1]
           * (float)phib[((size_t)b * N_DIM + n) * C_DIM + c1];
  float mean = block_reduce_sum(u0 + u1, sd) * (1.0f / C_DIM);
  float var  = block_reduce_sum(u0 * u0 + u1 * u1, sd) * (1.0f / C_DIM) - mean * mean;
  float inv = rsqrtf(var + 1e-6f);
  const size_t base = (size_t)bn * C_DIM;
  float v0 = (u0 - mean) * inv * w[c0] + bia[c0];
  float v1 = (u1 - mean) * inv * w[c1] + bia[c1];
  out1f[base + c0] = v0; out1f[base + c1] = v1;
  out1b[base + c0] = (bf16_t)v0; out1b[base + c1] = (bf16_t)v1;
}

// LN2 + relu + transpose to (b,c,n) into d_out
__global__ __launch_bounds__(256) void k_ln2(const float* __restrict__ o2,
                                             const float* __restrict__ w,
                                             const float* __restrict__ bia,
                                             float* __restrict__ out) {
  __shared__ float sd[256];
  const int bn = blockIdx.x;
  const int b = bn >> 12, n = bn & (N_DIM - 1);
  const int tid = threadIdx.x;
  const int c0 = tid, c1 = tid + 256;
  const size_t base = (size_t)bn * C_DIM;
  float u0 = o2[base + c0], u1 = o2[base + c1];
  float mean = block_reduce_sum(u0 + u1, sd) * (1.0f / C_DIM);
  float var  = block_reduce_sum(u0 * u0 + u1 * u1, sd) * (1.0f / C_DIM) - mean * mean;
  float inv = rsqrtf(var + 1e-6f);
  float v0 = (u0 - mean) * inv * w[c0] + bia[c0]; v0 = v0 > 0.f ? v0 : 0.f;
  float v1 = (u1 - mean) * inv * w[c1] + bia[c1]; v1 = v1 > 0.f ? v1 : 0.f;
  out[((size_t)b * C_DIM + c0) * N_DIM + n] = v0;
  out[((size_t)b * C_DIM + c1) * N_DIM + n] = v1;
}

// ---------------------------------------------------------------------------

extern "C" void kernel_launch(void* const* d_in, const int* in_sizes, int n_in,
                              void* d_out, int out_size, void* d_ws, size_t ws_size,
                              hipStream_t stream) {
  const float* x    = (const float*)d_in[0];
  const float* Wq   = (const float*)d_in[1];
  const float* Wk   = (const float*)d_in[2];
  const float* Wv   = (const float*)d_in[3];
  const float* Wphi = (const float*)d_in[4];
  const float* ln1w = (const float*)d_in[5];
  const float* ln1b = (const float*)d_in[6];
  const float* fc1w = (const float*)d_in[7];
  const float* fc1b = (const float*)d_in[8];
  const float* fc2w = (const float*)d_in[9];
  const float* fc2b = (const float*)d_in[10];
  const float* ln2w = (const float*)d_in[11];
  const float* ln2b = (const float*)d_in[12];
  float* out = (float*)d_out;

  char* p = (char*)d_ws;
  auto alloc = [&](size_t bytes) -> char* {
    char* r = p; p += (bytes + 255) & ~(size_t)255; return r;
  };

  bf16_t* xbT   = (bf16_t*)alloc((size_t)16 * N_DIM * C_DIM * 2);
  bf16_t* qt    = (bf16_t*)alloc((size_t)M_REST * N_DIM * C_DIM * 2);
  bf16_t* kb    = (bf16_t*)alloc((size_t)BB * C_DIM * N_DIM * 2);
  bf16_t* khat  = (bf16_t*)alloc((size_t)BB * C_DIM * N_DIM * 2);
  bf16_t* vb    = (bf16_t*)alloc((size_t)BB * C_DIM * N_DIM * 2);
  bf16_t* phib  = (bf16_t*)alloc((size_t)16 * N_DIM * C_DIM * 2);
  bf16_t* kvt   = (bf16_t*)alloc((size_t)BB * C_DIM * C_DIM * 2);
  bf16_t* wqb   = (bf16_t*)alloc((size_t)4 * CG * CG * 2);
  bf16_t* wkb   = (bf16_t*)alloc((size_t)4 * CG * CG * 2);
  bf16_t* wvb   = (bf16_t*)alloc((size_t)C_DIM * C_DIM * 2);
  bf16_t* wpb   = (bf16_t*)alloc((size_t)C_DIM * C_DIM * 2);
  bf16_t* fc1wb = (bf16_t*)alloc((size_t)H_DIM * C_DIM * 2);
  bf16_t* fc2wb = (bf16_t*)alloc((size_t)C_DIM * H_DIM * 2);
  float*  Qg    = (float*)alloc((size_t)BB * C_DIM * 4);
  float*  alphv = (float*)alloc((size_t)BB * N_DIM * 4);
  float*  ksum  = (float*)alloc((size_t)BB * C_DIM * 4);
  float*  zbuf  = (float*)alloc((size_t)M_REST * N_DIM * 4);
  float*  ymean = (float*)alloc((size_t)N_DIM * C_DIM * 4);
  float*  out1f = (float*)alloc((size_t)BB * N_DIM * C_DIM * 4);
  bf16_t* out1b = (bf16_t*)alloc((size_t)BB * N_DIM * C_DIM * 2);
  bf16_t* hbuf  = (bf16_t*)alloc((size_t)BB * N_DIM * H_DIM * 2);
  float*  o2    = (float*)alloc((size_t)BB * N_DIM * C_DIM * 4);

  // weight conversions to bf16
  k_cvt_bf16<<<64, 256, 0, stream>>>(Wq, wqb, 4 * CG * CG);
  k_cvt_bf16<<<64, 256, 0, stream>>>(Wk, wkb, 4 * CG * CG);
  k_cvt_bf16<<<256, 256, 0, stream>>>(Wv, wvb, C_DIM * C_DIM);
  k_cvt_bf16<<<256, 256, 0, stream>>>(Wphi, wpb, C_DIM * C_DIM);
  k_cvt_bf16<<<1024, 256, 0, stream>>>(fc1w, fc1wb, H_DIM * C_DIM);
  k_cvt_bf16<<<1024, 256, 0, stream>>>(fc2w, fc2wb, C_DIM * H_DIM);

  // x -> transposed bf16 (GEMM B operand layout)
  dim3 tb(32, 8), tg(C_DIM / 32, N_DIM / 32, 16);
  k_transpose_cvt<<<tg, tb, 0, stream>>>(x, xbT);

  // q / k projections (grouped 1x1) + phi     (16 * 16 * 128 macro-tiles)
  k_qk<<<(16 * CT32 * NT32) / 8, 256, 0, stream>>>(xbT, wqb, wkb, qt, kb);

  // Qg, alpha, khat/ksum
  k_qg<<<BB * C_DIM, 256, 0, stream>>>(kb, Qg);
  k_alpha<<<BB, 512, 0, stream>>>(kb, Qg, alphv);
  k_khat<<<BB * C_DIM, 256, 0, stream>>>(kb, alphv, khat, ksum);

  // v and phi projections
  k_vphi<<<(20 * CT32 * NT32) / 8, 256, 0, stream>>>(xbT, wvb, wpb, vb, phib);

  // kv outer product (only 4 distinct batches)
  k_kv<<<(BB * CT32 * CT32) / 8, 256, 0, stream>>>(khat, vb, kvt);

  // z normalizer
  k_z<<<(M_REST * N_DIM) / 8, 256, 0, stream>>>(qt, ksum, zbuf);

  // y = (kv^T q) * z * phi_rest, mean over m
  k_y<<<(CT32 * NT32) / 8, 256, 0, stream>>>(kvt, qt, zbuf, phib, ymean);

  // LN1 of residual
  k_ln1<<<BB * N_DIM, 256, 0, stream>>>(x, ymean, phib, ln1w, ln1b, out1f, out1b);

  // MLP
  k_fc1<<<(BB * HT32 * NT32) / 8, 256, 0, stream>>>(fc1wb, out1b, fc1b, hbuf);
  k_fc2<<<(BB * CT32 * NT32) / 8, 256, 0, stream>>>(fc2wb, hbuf, fc2b, out1f, o2);

  // LN2 + relu + transpose into output
  k_ln2<<<BB * N_DIM, 256, 0, stream>>>(o2, ln2w, ln2b, out);

  // rest passes through unchanged
  hipMemcpyAsync(out + (size_t)BB * C_DIM * N_DIM,
                 x + (size_t)BB * C_DIM * N_DIM,
                 (size_t)M_REST * C_DIM * N_DIM * sizeof(float),
                 hipMemcpyDeviceToDevice, stream);

  (void)in_sizes; (void)n_in; (void)out_size; (void)ws_size;
}
